// Model_7997229105644
// MI455X (gfx1250) — matrix operand order repaired and checked by reading, not compiled, not run on hardware
//
#include <hip/hip_runtime.h>

// ---------------------------------------------------------------------------
// Mamba-2 SSD forward for MI455X (gfx1250), wave32 + WMMA bf16.
// One workgroup (8 waves, 256 threads) per (batch, head); sequential loop over
// the 16 chunks carrying the 64x128 state in WMMA accumulator fragments.
// All exp() decay factors are precomputed per chunk: exp(a_l - a_s) is formed
// as eP[l]*eM[s] (safe in fp32: |a| < 64 -> factors within [1.6e-28, 6.2e27]).
// ---------------------------------------------------------------------------

typedef __attribute__((ext_vector_type(16))) __bf16 v16bf;
typedef __attribute__((ext_vector_type(8)))  float  v8f;

union FragU {
  v16bf v;
  uint4 q[2];
};

__device__ __forceinline__ unsigned short f2bf(float x) {
  unsigned int u = __float_as_uint(x);
  u += 0x7FFFu + ((u >> 16) & 1u);               // round-to-nearest-even
  return (unsigned short)(u >> 16);
}
__device__ __forceinline__ unsigned int pack2bf(float a, float b) {
  return (unsigned int)f2bf(a) | ((unsigned int)f2bf(b) << 16);
}

// A-matrix fragment (16x32 bf16, M = lane%16): per ISA table, lane half h2
// holds K = kb + 8*h2 + [0..7] and K = kb + 16 + 8*h2 + [0..7].
__device__ __forceinline__ v16bf loadA(const unsigned short* base, int row,
                                       int kb, int stride, int h2) {
  FragU f;
  const unsigned short* r = base + row * stride + kb + 8 * h2;
  f.q[0] = *(const uint4*)(r);
  f.q[1] = *(const uint4*)(r + 16);
  return f.v;
}

// B-matrix fragment (32x16 bf16, N = lane%16): lane half h2 holds
// K = kb + 8*h2 + [0..7] and K = kb + 16 + 8*h2 + [0..7]. LDS layout must be [N][K].
__device__ __forceinline__ v16bf loadB(const unsigned short* base, int row,
                                       int kb, int stride, int h2) {
  FragU f;
  const unsigned short* r = base + row * stride + kb + 8 * h2;
  f.q[0] = *(const uint4*)(r);
  f.q[1] = *(const uint4*)(r + 16);
  return f.v;
}

#define WMMA_BF16(A, B, C)                                                    \
  __builtin_amdgcn_wmma_f32_16x16x32_bf16(false, (A), false, (B), (short)0,   \
                                          (C), false, false)

// Problem constants (from reference setup_inputs).
#define DB 16
#define DS 1024
#define DH 32
#define DP 64
#define DN 128
#define CL 64     // chunk length
#define NC 16     // chunks

// Padded LDS row strides (bf16 elements); +8 keeps ds_load_b128 conflict-free.
#define STN 136   // for K = 128 tiles
#define STL 72    // for K = 64 tiles

__global__ void __launch_bounds__(256)
ssd_fwd_kernel(const float* __restrict__ Xg, const float* __restrict__ Ag,
               const float* __restrict__ Bg, const float* __restrict__ Cg,
               float* __restrict__ Yg) {
  extern __shared__ __align__(16) char smem[];
  unsigned short* sC  = (unsigned short*)smem;   // [64][136] C chunk  (A-mat M1,M3)
  unsigned short* sB  = sC  + CL * STN;          // [64][136] B chunk  (B-mat M1, [s][n])
  unsigned short* sBt = sB  + CL * STN;          // [128][72] B^T * decay (B-mat M4, [n][l])
  unsigned short* sXt = sBt + DN * STL;          // [64][72]  X^T       (B-mat M2 / A-mat M4, [p][l])
  unsigned short* sG  = sXt + CL * STL;          // [64][72]  masked G  (A-mat M2, [l][s])
  unsigned short* sSb = sG  + CL * STL;          // [64][136] state bf16 (B-mat M3, [p][n])
  float* aCum = (float*)(sSb + CL * STN);        // [64] inclusive cumsum of A
  float* eP   = aCum + CL;                       // [64] exp(+aCum[l])
  float* eM   = eP + CL;                         // [64] exp(-aCum[l])

  const int tid  = threadIdx.x;
  const int w    = tid >> 5;
  const int lane = tid & 31;
  const int h2   = lane >> 4;     // lane half
  const int nl   = lane & 15;     // position within half
  const int bb   = blockIdx.x >> 5;
  const int hh   = blockIdx.x & 31;

  // Tile ownership (shared across M1 / M3+M2): wave w covers M-tile w>>1 and
  // two N-tiles (w&1)*2 + {0,1}.
  const int mt  = w >> 1;
  const int ntB = (w & 1) * 2;
  // State tiles: S is 64(p) x 128(n) => wave w owns p-tile (w&3) and n-tiles
  // (w>>2)*64 + j*16, j=0..3.
  const int pt_s = w & 3;
  const int ng_s = w >> 2;
  v8f accS[4];
#pragma unroll
  for (int j = 0; j < 4; ++j) accS[j] = (v8f){};

  for (int c = 0; c < NC; ++c) {
    const int s0 = c * CL;

    // ---- stage 1: A chunk -> LDS, serial fp32 cumsum, exp tables ----
    if (tid < CL)
      aCum[tid] = Ag[(size_t)((bb * DS + s0 + tid) * DH) + hh];
    __syncthreads();
    if (tid == 0) {
      float run = 0.f;
      for (int l = 0; l < CL; ++l) { run += aCum[l]; aCum[l] = run; }
    }
    __syncthreads();
    if (tid < CL) {
      float a = aCum[tid];
      eP[tid] = expf(a);
      eM[tid] = expf(-a);
    }
    __syncthreads();

    const float eLast = eP[CL - 1];              // exp(a_last)

    // ---- stage 2: global -> LDS (bf16), plus state dump S -> sSb ----
    for (int base = tid * 4; base < CL * DN; base += 1024) {
      int l = base >> 7, n = base & 127;
      const float4 cv = *(const float4*)&Cg[((bb * DS + s0 + l) * DH + hh) * DN + n];
      const float4 bv = *(const float4*)&Bg[((bb * DS + s0 + l) * DH + hh) * DN + n];
      *(unsigned int*)&sC[l * STN + n]     = pack2bf(cv.x, cv.y);
      *(unsigned int*)&sC[l * STN + n + 2] = pack2bf(cv.z, cv.w);
      *(unsigned int*)&sB[l * STN + n]     = pack2bf(bv.x, bv.y);
      *(unsigned int*)&sB[l * STN + n + 2] = pack2bf(bv.z, bv.w);
      const float dec = eLast * eM[l];           // exp(a_last - a_l)
      sBt[(n + 0) * STL + l] = f2bf(bv.x * dec);
      sBt[(n + 1) * STL + l] = f2bf(bv.y * dec);
      sBt[(n + 2) * STL + l] = f2bf(bv.z * dec);
      sBt[(n + 3) * STL + l] = f2bf(bv.w * dec);
    }
    for (int base = tid * 4; base < CL * DP; base += 1024) {
      int l = base >> 6, pp = base & 63;
      const float4 xv = *(const float4*)&Xg[((bb * DS + s0 + l) * DH + hh) * DP + pp];
      sXt[(pp + 0) * STL + l] = f2bf(xv.x);
      sXt[(pp + 1) * STL + l] = f2bf(xv.y);
      sXt[(pp + 2) * STL + l] = f2bf(xv.z);
      sXt[(pp + 3) * STL + l] = f2bf(xv.w);
    }
    // dump carried state (pre-update, = states_in for this chunk) as bf16
#pragma unroll
    for (int j = 0; j < 4; ++j) {
#pragma unroll
      for (int v = 0; v < 8; ++v) {
        int p = pt_s * 16 + v + 8 * h2;
        int n = ng_s * 64 + j * 16 + nl;
        sSb[p * STN + n] = f2bf(accS[j][v]);
      }
    }
    __syncthreads();

    // prefetch next chunk's B/C rows into cache while we compute
    if (c + 1 < NC) {
      int l = tid >> 2, n = (tid & 3) * 32;
      int g = ((bb * DS + s0 + CL + l) * DH + hh) * DN + n;
      __builtin_prefetch(&Cg[g], 0, 1);
      __builtin_prefetch(&Bg[g], 0, 1);
    }

    // ---- stage 3 (M1): G = C * B^T (64x64, K=128); mask with L; -> sG ----
    {
      v8f acc0 = (v8f){}, acc1 = (v8f){};
#pragma unroll
      for (int kb = 0; kb < DN; kb += 32) {
        v16bf af = loadA(sC, mt * 16 + nl, kb, STN, h2);   // shared A-frag
        acc0 = WMMA_BF16(af, loadB(sB, (ntB + 0) * 16 + nl, kb, STN, h2), acc0);
        acc1 = WMMA_BF16(af, loadB(sB, (ntB + 1) * 16 + nl, kb, STN, h2), acc1);
      }
      const int ss0 = ntB * 16 + nl, ss1 = ss0 + 16;
      const float em0 = eM[ss0], em1 = eM[ss1];
#pragma unroll
      for (int v = 0; v < 8; ++v) {
        int l = mt * 16 + v + 8 * h2;
        float epl = eP[l];
        sG[l * STL + ss0] = f2bf((l >= ss0) ? acc0[v] * epl * em0 : 0.0f);
        sG[l * STL + ss1] = f2bf((l >= ss1) ? acc1[v] * epl * em1 : 0.0f);
      }
    }
    __syncthreads();

    // ---- stage 4 (M3 + M2): Y = exp(a_l) * (C @ S^T)  +  G_masked @ X ----
    {
      v8f acc0 = (v8f){}, acc1 = (v8f){};
#pragma unroll
      for (int kb = 0; kb < DN; kb += 32) {     // K over n
        v16bf af = loadA(sC, mt * 16 + nl, kb, STN, h2);
        acc0 = WMMA_BF16(af, loadB(sSb, (ntB + 0) * 16 + nl, kb, STN, h2), acc0);
        acc1 = WMMA_BF16(af, loadB(sSb, (ntB + 1) * 16 + nl, kb, STN, h2), acc1);
      }
#pragma unroll
      for (int v = 0; v < 8; ++v) {             // state_decay_out = exp(a_l)
        float sc = eP[mt * 16 + v + 8 * h2];
        acc0[v] *= sc;
        acc1[v] *= sc;
      }
#pragma unroll
      for (int kb = 0; kb < CL; kb += 32) {     // K over s
        v16bf af = loadA(sG, mt * 16 + nl, kb, STL, h2);
        acc0 = WMMA_BF16(af, loadB(sXt, (ntB + 0) * 16 + nl, kb, STL, h2), acc0);
        acc1 = WMMA_BF16(af, loadB(sXt, (ntB + 1) * 16 + nl, kb, STL, h2), acc1);
      }
#pragma unroll
      for (int v = 0; v < 8; ++v) {
        int l = mt * 16 + v + 8 * h2;
        float* row = Yg + (size_t)((bb * DS + s0 + l) * DH + hh) * DP;
        __builtin_nontemporal_store(acc0[v], row + (ntB + 0) * 16 + nl);
        __builtin_nontemporal_store(acc1[v], row + (ntB + 1) * 16 + nl);
      }
    }

    // ---- stage 5 (M4): S = exp(a_last)*S + X^T @ (B * exp(a_last-a_l)) ----
    {
#pragma unroll
      for (int j = 0; j < 4; ++j)
#pragma unroll
        for (int v = 0; v < 8; ++v) accS[j][v] *= eLast;
#pragma unroll
      for (int kb = 0; kb < CL; kb += 32) {     // K over l
        v16bf af = loadA(sXt, pt_s * 16 + nl, kb, STL, h2);  // shared A-frag
#pragma unroll
        for (int j = 0; j < 4; ++j) {
          v16bf bf = loadB(sBt, ng_s * 64 + j * 16 + nl, kb, STL, h2);
          accS[j] = WMMA_BF16(af, bf, accS[j]);
        }
      }
    }
    __syncthreads();   // protect LDS before next chunk overwrites it
  }
}

extern "C" void kernel_launch(void* const* d_in, const int* in_sizes, int n_in,
                              void* d_out, int out_size, void* d_ws,
                              size_t ws_size, hipStream_t stream) {
  const float* X = (const float*)d_in[0];
  const float* A = (const float*)d_in[1];
  const float* B = (const float*)d_in[2];
  const float* C = (const float*)d_in[3];
  float* Y = (float*)d_out;
  (void)in_sizes; (void)n_in; (void)out_size; (void)d_ws; (void)ws_size;

  // LDS: 3*(64*136) + 128*72 + 2*(64*72) bf16 + 3*64 floats ~= 89.9 KB / WG
  const size_t shmem = 3 * (CL * STN) * 2 + (DN * STL) * 2 +
                       2 * (CL * STL) * 2 + (3 * CL) * sizeof(float);
  (void)hipFuncSetAttribute((const void*)ssd_fwd_kernel,
                            hipFuncAttributeMaxDynamicSharedMemorySize,
                            (int)shmem);
  dim3 grid(DB * DH);   // 512 independent (batch, head) streams
  dim3 block(256);      // 8 wave32 waves
  ssd_fwd_kernel<<<grid, block, shmem, stream>>>(X, A, B, C, Y);
}
